// SelectiveSSM_61821759259048
// MI455X (gfx1250) — matrix operand order repaired and checked by reading, not compiled, not run on hardware
//
#include <hip/hip_runtime.h>
#include <stdint.h>

#define BATCH   2
#define SEQ     2048
#define HDIM    1024
#define INNER   2048
#define STATE   16
#define DT_RANK 64
#define XDBL    96            // DT_RANK + 2*STATE
#define MROWS   (BATCH*SEQ)   // 4096
#define KERN    4

typedef __bf16 bf16_t;
typedef __attribute__((ext_vector_type(16))) __bf16 v16bf;
typedef __attribute__((ext_vector_type(8)))  __bf16 v8bf;
typedef __attribute__((ext_vector_type(8)))  float  v8f;

// ---------------------------------------------------------------- helpers ---
__device__ __forceinline__ bf16_t f2bf(float f) {
  uint32_t u = __float_as_uint(f);
  u += 0x7FFFu + ((u >> 16) & 1u);          // round-to-nearest-even
  uint16_t h = (uint16_t)(u >> 16);
  bf16_t b;
  __builtin_memcpy(&b, &h, 2);
  return b;
}

__device__ __forceinline__ float silu_f(float v) {
  return v / (1.0f + __expf(-v));
}

// ------------------------------------------------------------ f32 -> bf16 ---
__global__ void k_cvt_bf16(const float* __restrict__ src,
                           bf16_t* __restrict__ dst, int n) {
  int i = blockIdx.x * blockDim.x + threadIdx.x;
  if (i < n) dst[i] = f2bf(src[i]);
}

// ------------------------------------------------- WMMA NT GEMM (bf16->f32) -
// C[M,N] = A[M,K] * B[N,K]^T ; one wave per block; wave tile = (MT*16)x(NT*16)
template <int MT, int NT>
__global__ void __launch_bounds__(32)
k_gemm_nt(const bf16_t* __restrict__ A, int lda,
          const bf16_t* __restrict__ Bm, int ldb,
          float* __restrict__ C, int ldc, int K) {
  const int lane = threadIdx.x;
  const int half = lane >> 4;     // 0: lanes 0-15, 1: lanes 16-31
  const int l15  = lane & 15;
  const int m0   = blockIdx.x * (MT * 16);
  const int n0   = blockIdx.y * (NT * 16);

  v8f acc[MT][NT];
#pragma unroll
  for (int i = 0; i < MT; ++i)
#pragma unroll
    for (int j = 0; j < NT; ++j)
      acc[i][j] = {};

  const int a_k = half * 8;       // A frag: K[a_k..a_k+7] then K[a_k+16..a_k+23]
  const int b_k = half * 8;       // B frag: K[b_k..b_k+7] then K[b_k+16..b_k+23]

  for (int k = 0; k < K; k += 32) {
    v16bf af[MT];
#pragma unroll
    for (int i = 0; i < MT; ++i) {
      const bf16_t* ap = A + (size_t)(m0 + i * 16 + l15) * lda + k + a_k;
      v8bf lo = *(const v8bf*)(ap);
      v8bf hi = *(const v8bf*)(ap + 16);
      af[i] = __builtin_shufflevector(lo, hi, 0, 1, 2, 3, 4, 5, 6, 7,
                                               8, 9, 10, 11, 12, 13, 14, 15);
    }
    v16bf bfv[NT];
#pragma unroll
    for (int j = 0; j < NT; ++j) {
      const bf16_t* bp = Bm + (size_t)(n0 + j * 16 + l15) * ldb + k + b_k;
      v8bf lo = *(const v8bf*)(bp);
      v8bf hi = *(const v8bf*)(bp + 16);
      bfv[j] = __builtin_shufflevector(lo, hi, 0, 1, 2, 3, 4, 5, 6, 7,
                                                8, 9, 10, 11, 12, 13, 14, 15);
    }
    // single uniform branch; near-scope speculative prefetch of next K block
    if (k + 32 < K) {
#pragma unroll
      for (int i = 0; i < MT; ++i)
        __builtin_prefetch(A + (size_t)(m0 + i * 16 + l15) * lda +
                               (k + 32 + a_k), 0, 3);
#pragma unroll
      for (int j = 0; j < NT; ++j)
        __builtin_prefetch(Bm + (size_t)(n0 + j * 16 + l15) * ldb +
                               (k + 32 + b_k), 0, 3);
    }
#pragma unroll
    for (int i = 0; i < MT; ++i)
#pragma unroll
      for (int j = 0; j < NT; ++j)
        acc[i][j] = __builtin_amdgcn_wmma_f32_16x16x32_bf16(
            false, af[i], false, bfv[j], (short)0, acc[i][j], false, false);
  }

  // C/D layout: VGPR r, lanes 0-15 -> (M=r, N=lane); lanes 16-31 -> (M=r+8)
#pragma unroll
  for (int i = 0; i < MT; ++i)
#pragma unroll
    for (int j = 0; j < NT; ++j) {
      const int col = n0 + j * 16 + l15;
#pragma unroll
      for (int r = 0; r < 8; ++r) {
        const int row = m0 + i * 16 + half * 8 + r;
        C[(size_t)row * ldc + col] = acc[i][j][r];
      }
    }
}

// ------------------------------------- depthwise causal conv(4)+bias+SiLU ---
__global__ void __launch_bounds__(256)
k_conv_silu(const float* __restrict__ xz, const float* __restrict__ conv_w,
            const float* __restrict__ conv_b,
            float* __restrict__ xin, bf16_t* __restrict__ xin_bf) {
  int i = blockIdx.x * 256 + threadIdx.x;          // over (b,t,c)
  int c = i & (INNER - 1);
  int t = (i >> 11) & (SEQ - 1);
  int b = i >> 22;
  float acc = conv_b[c];
#pragma unroll
  for (int kk = 0; kk < KERN; ++kk) {
    int tt = t - (KERN - 1) + kk;
    if (tt >= 0)
      acc += conv_w[c * KERN + kk] *
             xz[(((size_t)b * SEQ + tt) << 12) + c]; // xz row stride 4096
  }
  float s = silu_f(acc);
  xin[i]    = s;
  xin_bf[i] = f2bf(s);
}

// --------------------------------------------------- dt = softplus(v + b) ---
__global__ void __launch_bounds__(256)
k_dt_softplus(float* __restrict__ dt, const float* __restrict__ b_dt) {
  int i = blockIdx.x * 256 + threadIdx.x;
  int c = i & (INNER - 1);
  float v = dt[i] + b_dt[c];
  dt[i] = (v > 20.0f) ? v : log1pf(__expf(v));
}

// ------------------------------------------------------- selective scan -----
#define TCHUNK 128
__global__ void __launch_bounds__(256)
k_scan(const float* __restrict__ xin, const float* __restrict__ dt,
       const float* __restrict__ x_dbl, const float* __restrict__ A_log,
       const float* __restrict__ state0,
       float* __restrict__ y, float* __restrict__ fstate) {
  __shared__ float sBC[TCHUNK * 2 * STATE];        // 16 KB
  const int b = blockIdx.y;
  const int c = blockIdx.x * 256 + threadIdx.x;    // channel 0..INNER-1

  float Av[STATE], st[STATE];
#pragma unroll
  for (int s = 0; s < STATE; ++s) {
    Av[s] = -__expf(A_log[c * STATE + s]);
    st[s] = state0[(((size_t)b * INNER) + c) * STATE + s];
  }

  const size_t base   = ((size_t)b * SEQ) * INNER + c;
  const size_t xdbase = ((size_t)b * SEQ) * XDBL;

  for (int t0 = 0; t0 < SEQ; t0 += TCHUNK) {
    __syncthreads();
    for (int i = threadIdx.x; i < TCHUNK * 32; i += 256) {
      int tl = i >> 5, s = i & 31;                 // s<16 -> B, s>=16 -> C
      sBC[i] = x_dbl[xdbase + (size_t)(t0 + tl) * XDBL + DT_RANK + s];
    }
    __syncthreads();
    for (int tl = 0; tl < TCHUNK; ++tl) {
      const int t   = t0 + tl;
      const float d = dt[base + (size_t)t * INNER];
      const float x = xin[base + (size_t)t * INNER];
      const float dx = d * x;
      const float* BC = &sBC[tl * 32];
      float yv = 0.0f;
#pragma unroll
      for (int s = 0; s < STATE; ++s) {
        float dA = __expf(d * Av[s]);
        st[s] = dA * st[s] + dx * BC[s];
        yv += st[s] * BC[STATE + s];
      }
      y[base + (size_t)t * INNER] = yv;
    }
  }
#pragma unroll
  for (int s = 0; s < STATE; ++s)
    fstate[(((size_t)b * INNER) + c) * STATE + s] = st[s];
}

// ------------------------------------------------ g = (y + xin*D)*silu(z) ---
__global__ void __launch_bounds__(256)
k_gate(const float* __restrict__ y, const float* __restrict__ xin,
       const float* __restrict__ xz, const float* __restrict__ Dv,
       bf16_t* __restrict__ g) {
  int i = blockIdx.x * 256 + threadIdx.x;
  int c = i & (INNER - 1);
  size_t row = (size_t)(i >> 11);                  // b*SEQ + t
  float z = xz[(row << 12) + INNER + c];
  float v = (y[i] + xin[i] * Dv[c]) * silu_f(z);
  g[i] = f2bf(v);
}

// ---------------------------------------------------------------- launch ----
extern "C" void kernel_launch(void* const* d_in, const int* in_sizes, int n_in,
                              void* d_out, int out_size, void* d_ws,
                              size_t ws_size, hipStream_t stream) {
  const float* x      = (const float*)d_in[0];
  const float* state0 = (const float*)d_in[1];
  const float* W_in   = (const float*)d_in[2];
  const float* conv_w = (const float*)d_in[3];
  const float* conv_b = (const float*)d_in[4];
  const float* W_x    = (const float*)d_in[5];
  const float* W_dt   = (const float*)d_in[6];
  const float* b_dt   = (const float*)d_in[7];
  const float* A_log  = (const float*)d_in[8];
  const float* Dv     = (const float*)d_in[9];
  const float* W_out  = (const float*)d_in[10];

  float* out    = (float*)d_out;                       // (B,L,H)
  float* fstate = out + (size_t)BATCH * SEQ * HDIM;    // (B,INNER,STATE)

  char* ws = (char*)d_ws;
  size_t off = 0;
  auto carve = [&](size_t bytes) -> void* {
    void* p = ws + off;
    off += (bytes + 255) & ~(size_t)255;
    return p;
  };
  float*  xz      = (float*) carve((size_t)MROWS * 4096 * 4);
  float*  xin     = (float*) carve((size_t)MROWS * INNER * 4);
  float*  dtb     = (float*) carve((size_t)MROWS * INNER * 4);
  float*  yb      = (float*) carve((size_t)MROWS * INNER * 4);
  float*  xdbl    = (float*) carve((size_t)MROWS * XDBL * 4);
  bf16_t* x_bf    = (bf16_t*)carve((size_t)MROWS * HDIM * 2);
  bf16_t* Win_bf  = (bf16_t*)carve((size_t)4096 * HDIM * 2);
  bf16_t* Wx_bf   = (bf16_t*)carve((size_t)XDBL * INNER * 2);
  bf16_t* Wdt_bf  = (bf16_t*)carve((size_t)INNER * DT_RANK * 2);
  bf16_t* Wout_bf = (bf16_t*)carve((size_t)HDIM * INNER * 2);
  bf16_t* xin_bf  = (bf16_t*)carve((size_t)MROWS * INNER * 2);
  bf16_t* xdbl_bf = (bf16_t*)carve((size_t)MROWS * XDBL * 2);
  bf16_t* g_bf    = (bf16_t*)carve((size_t)MROWS * INNER * 2);

  auto cvt = [&](const float* s, bf16_t* d, size_t n) {
    k_cvt_bf16<<<dim3((unsigned)((n + 255) / 256)), dim3(256), 0, stream>>>(
        s, d, (int)n);
  };
  cvt(x,     x_bf,    (size_t)MROWS * HDIM);
  cvt(W_in,  Win_bf,  (size_t)4096 * HDIM);
  cvt(W_x,   Wx_bf,   (size_t)XDBL * INNER);
  cvt(W_dt,  Wdt_bf,  (size_t)INNER * DT_RANK);
  cvt(W_out, Wout_bf, (size_t)HDIM * INNER);

  // xz = x @ W_in^T : (4096,1024)x(4096,1024)^T -> (4096,4096)
  k_gemm_nt<4, 4><<<dim3(MROWS / 64, 4096 / 64), dim3(32), 0, stream>>>(
      x_bf, HDIM, Win_bf, HDIM, xz, 4096, HDIM);

  // depthwise conv + SiLU
  k_conv_silu<<<dim3((MROWS * INNER) / 256), dim3(256), 0, stream>>>(
      xz, conv_w, conv_b, xin, xin_bf);

  // x_dbl = xin @ W_x^T : (4096,2048)x(96,2048)^T -> (4096,96)
  k_gemm_nt<4, 2><<<dim3(MROWS / 64, XDBL / 32), dim3(32), 0, stream>>>(
      xin_bf, INNER, Wx_bf, INNER, xdbl, XDBL, INNER);

  cvt(xdbl, xdbl_bf, (size_t)MROWS * XDBL);

  // dt_lin = x_dbl[:, :64] @ W_dt^T : (4096,64)x(2048,64)^T -> (4096,2048)
  k_gemm_nt<4, 4><<<dim3(MROWS / 64, INNER / 64), dim3(32), 0, stream>>>(
      xdbl_bf, XDBL, Wdt_bf, DT_RANK, dtb, INNER, DT_RANK);

  k_dt_softplus<<<dim3((MROWS * INNER) / 256), dim3(256), 0, stream>>>(
      dtb, b_dt);

  // sequential selective scan; also writes final_state output region
  k_scan<<<dim3(INNER / 256, BATCH), dim3(256), 0, stream>>>(
      xin, dtb, xdbl, A_log, state0, yb, fstate);

  k_gate<<<dim3((MROWS * INNER) / 256), dim3(256), 0, stream>>>(
      yb, xin, xz, Dv, g_bf);

  // out = g @ W_out^T : (4096,2048)x(1024,2048)^T -> (4096,1024)
  k_gemm_nt<4, 4><<<dim3(MROWS / 64, HDIM / 64), dim3(32), 0, stream>>>(
      g_bf, INNER, Wout_bf, INNER, out, HDIM, INNER);
}
